// critic_attention_v3_49409303773227
// MI455X (gfx1250) — matrix operand order repaired and checked by reading, not compiled, not run on hardware
//
#include <hip/hip_runtime.h>
#include <hip/hip_bf16.h>

// ---------------------------------------------------------------------------
// MARL critic-with-attention, fully WMMA-ified for gfx1250 (MI455X).
//   All four GEMMs (MLP 24->32, V 32->16, Wc1 48->256, Wc2 256->256) run on
//   v_wmma_f32_16x16x32_f16 out of LDS. Attention logits reduced to two
//   precomputed 32-vectors (Wq/Wk/Wa folded). GEMM1/GEMM2 process two
//   N-tiles concurrently (independent accumulator chains) to keep the
//   matrix pipe busy at low occupancy. One __syncthreads total: every wave
//   reads only LDS rows it wrote itself.
// ---------------------------------------------------------------------------

typedef __attribute__((ext_vector_type(16))) _Float16 v16h;
typedef __attribute__((ext_vector_type(8)))  float    v8f;

union F16Frag { v16h v; uint4 u[2]; };

__device__ inline v16h lds_frag(const _Float16* p0, const _Float16* p1) {
    F16Frag f;
    f.u[0] = *(const uint4*)p0;
    f.u[1] = *(const uint4*)p1;
    return f.v;
}

__device__ inline float lrelu(float x) { return x > 0.f ? x : 0.01f * x; }

#define WMMA_F16(A, B, C) \
    __builtin_amdgcn_wmma_f32_16x16x32_f16(false, (A), false, (B), (short)0, (C), false, false)

// ---------------------------------------------------------------------------
// Prep: f16 weight images + folded attention vectors into d_ws.
// ---------------------------------------------------------------------------
__global__ void critic_prep(const float* __restrict__ Wc2, const float* __restrict__ Wc1,
                            const float* __restrict__ W_mlp, const float* __restrict__ Wv,
                            const float* __restrict__ Wk, const float* __restrict__ Wq,
                            const float* __restrict__ Wa, const float* __restrict__ bk,
                            const float* __restrict__ bq, const float* __restrict__ ba,
                            _Float16* __restrict__ w2h, _Float16* __restrict__ w1h,
                            _Float16* __restrict__ wmlph, _Float16* __restrict__ wvh,
                            float* __restrict__ wka, float* __restrict__ wqa,
                            float* __restrict__ consta) {
    int i = blockIdx.x * 256 + threadIdx.x;
    if (i < 256 * 256) w2h[i] = (_Float16)Wc2[i];
    if (i < 256 * 64) {
        int o = i >> 6, c = i & 63;
        w1h[i] = (c < 48) ? (_Float16)Wc1[o * 48 + c] : (_Float16)0.f;
    }
    if (i < 32 * 32) {
        int j = i >> 5, f = i & 31;
        wmlph[i] = (f < 24) ? (_Float16)W_mlp[j * 24 + f] : (_Float16)0.f;
    }
    if (i < 16 * 32) wvh[i] = (_Float16)Wv[i];
    if (i < 32) {
        float sk = 0.f, sq = 0.f;
        for (int d = 0; d < 16; ++d) {
            sk = fmaf(Wk[d * 32 + i], Wa[16 + d], sk);
            sq = fmaf(Wq[d * 32 + i], Wa[d], sq);
        }
        wka[i] = sk;
        wqa[i] = sq;
    }
    if (i == 0) {
        float c = ba[0];
        for (int d = 0; d < 16; ++d) c += bk[d] * Wa[16 + d] + bq[d] * Wa[d];
        *consta = c;
    }
}

// ---------------------------------------------------------------------------
// Main fused kernel: 128 samples/block, 4 waves, wave owns 32 samples
// (= 96 agent-rows = MLP/V tiles 6w..6w+5, GEMM1/2 tiles 2w..2w+1).
// ---------------------------------------------------------------------------
#define TB 128

__global__ __launch_bounds__(TB, 1)
void critic_attention_fused(
    const float* __restrict__ obs, const float* __restrict__ act,
    const float* __restrict__ b_mlp, const float* __restrict__ bv,
    const float* __restrict__ bc1, const float* __restrict__ bc2,
    const float* __restrict__ Wc, const float* __restrict__ bc,
    const float* __restrict__ wka, const float* __restrict__ wqa,
    const float* __restrict__ consta,
    const _Float16* __restrict__ w2h, const _Float16* __restrict__ w1h,
    const _Float16* __restrict__ wmlph, const _Float16* __restrict__ wvh,
    float* __restrict__ out) {

    extern __shared__ char smem_raw[];
    _Float16* sW2   = (_Float16*)smem_raw;       // [256][256]  128 KB
    _Float16* sW1   = sW2 + 256 * 256;           // [256][64]    32 KB
    _Float16* sWmlp = sW1 + 256 * 64;            // [32][32]      2 KB
    _Float16* sWv   = sWmlp + 32 * 32;           // [16][32]      1 KB
    _Float16* sX    = sWv + 16 * 32;             // [384][32]    24 KB  (MLP input)
    _Float16* sH    = sX + 384 * 32;             // [384][32]    24 KB  (h, f16)
    _Float16* sVall = sH + 384 * 32;             // [128][64]    16 KB  (v_all, K-pad)
    _Float16* sX1   = sVall + TB * 64;           // [128][256]   64 KB  (x1, f16)
    float*    sAttn = (float*)(sX1 + TB * 256);  // [384]        1.5 KB
    float*    sWc   = sAttn + 384;               // [256]        1 KB
    // total 300544 B < 320 KB

    const int t = threadIdx.x;

    // ---- P0: cooperative LDS fill of all weight images ----
    {
        const uint4* s2 = (const uint4*)w2h;
        uint4* d2 = (uint4*)sW2;
        for (int i = t; i < 8192; i += TB) d2[i] = s2[i];
        const uint4* s1 = (const uint4*)w1h;
        uint4* d1 = (uint4*)sW1;
        for (int i = t; i < 2048; i += TB) d1[i] = s1[i];
        const uint4* sm = (const uint4*)wmlph;
        uint4* dm = (uint4*)sWmlp;
        for (int i = t; i < 128; i += TB) dm[i] = sm[i];
        const uint4* sv = (const uint4*)wvh;
        uint4* dv = (uint4*)sWv;
        if (t < 64) dv[t] = sv[t];
        for (int i = t; i < 256; i += TB) sWc[i] = Wc[i];
    }

    // ---- P1: stage inputs as f16 rows [3 agents x 32(K-pad)] per sample ----
    {
        const size_t s = (size_t)blockIdx.x * TB + t;
        const float* ob = obs + s * 48;
        const float* ac = act + s * 24;
        _Float16* xr = sX + t * 96;            // rows 3t..3t+2, contiguous
        #pragma unroll
        for (int n = 0; n < 3; ++n) {
            #pragma unroll
            for (int f = 0; f < 16; ++f) xr[n * 32 + f] = (_Float16)ob[n * 16 + f];
            #pragma unroll
            for (int f = 0; f < 8; ++f)  xr[n * 32 + 16 + f] = (_Float16)ac[n * 8 + f];
            #pragma unroll
            for (int f = 24; f < 32; ++f) xr[n * 32 + f] = (_Float16)0.f;
        }
        _Float16* vr = sVall + t * 64;         // zero K-pad 48..63 once
        #pragma unroll
        for (int i = 48; i < 64; ++i) vr[i] = (_Float16)0.f;
    }

    __syncthreads();   // weights visible to all waves; all later deps are intra-wave

    const int lane   = t & 31;
    const int wave   = t >> 5;
    const int ln     = lane & 15;
    const int laneHi = lane >> 4;

    // ---- P2: GEMM-MLP  x[384x32] x W_mlp^T[32x32] -> h (lrelu, f16, LDS) ----
    for (int mi = 0; mi < 6; ++mi) {
        const int mt = wave * 6 + mi;          // rows mt*16..mt*16+15
        const _Float16* arow = sX + (size_t)(mt * 16 + ln) * 32;
        v16h A = lds_frag(arow + laneHi * 16, arow + laneHi * 16 + 8);
        // two N-tiles, independent accumulators
        v8f acc0, acc1;
        {
            float b0 = b_mlp[ln], b1 = b_mlp[16 + ln];
            #pragma unroll
            for (int r = 0; r < 8; ++r) { acc0[r] = b0; acc1[r] = b1; }
        }
        const _Float16* br0 = sWmlp + (size_t)ln * 32;
        const _Float16* br1 = sWmlp + (size_t)(16 + ln) * 32;
        v16h B0 = lds_frag(br0 + laneHi * 16, br0 + laneHi * 16 + 8);
        v16h B1 = lds_frag(br1 + laneHi * 16, br1 + laneHi * 16 + 8);
        acc0 = WMMA_F16(A, B0, acc0);
        acc1 = WMMA_F16(A, B1, acc1);
        const int m0 = mt * 16 + laneHi * 8;
        #pragma unroll
        for (int r = 0; r < 8; ++r) {
            sH[(size_t)(m0 + r) * 32 + ln]      = (_Float16)lrelu(acc0[r]);
            sH[(size_t)(m0 + r) * 32 + 16 + ln] = (_Float16)lrelu(acc1[r]);
        }
    }

    // ---- P3: attention logits + softmax (per sample, ~130 FMA) ----
    {
        _Float16 hbuf[96];
        #pragma unroll
        for (int c = 0; c < 12; ++c)
            ((uint4*)hbuf)[c] = *(const uint4*)(sH + (size_t)t * 96 + c * 8);

        float lg[3];
        float qd = 0.f;
        #pragma unroll
        for (int n = 0; n < 3; ++n) {
            float d = 0.f;
            #pragma unroll
            for (int j = 0; j < 32; ++j) {
                float hv = (float)hbuf[n * 32 + j];
                d = fmaf(hv, wka[j], d);
                if (n == 0) qd = fmaf(hv, wqa[j], qd);
            }
            lg[n] = d;
        }
        const float ca = *consta;
        #pragma unroll
        for (int n = 0; n < 3; ++n) lg[n] = (lg[n] + qd + ca) * 0.25f;  // / sqrt(16)
        float mx = fmaxf(lg[0], fmaxf(lg[1], lg[2]));
        float e0 = __expf(lg[0] - mx), e1 = __expf(lg[1] - mx), e2 = __expf(lg[2] - mx);
        float inv = 1.f / (e0 + e1 + e2);
        sAttn[t * 3 + 0] = e0 * inv;
        sAttn[t * 3 + 1] = e1 * inv;
        sAttn[t * 3 + 2] = e2 * inv;
    }

    // ---- P4: GEMM-V  h[384x32] x Wv^T[32x16], attn-scale, scatter -> v_all ----
    for (int mi = 0; mi < 6; ++mi) {
        const int mt = wave * 6 + mi;
        const _Float16* arow = sH + (size_t)(mt * 16 + ln) * 32;
        v16h A = lds_frag(arow + laneHi * 16, arow + laneHi * 16 + 8);
        float bias = bv[ln];
        v8f acc;
        #pragma unroll
        for (int r = 0; r < 8; ++r) acc[r] = bias;
        const _Float16* brow = sWv + (size_t)ln * 32;
        v16h Bf = lds_frag(brow + laneHi * 16, brow + laneHi * 16 + 8);
        acc = WMMA_F16(A, Bf, acc);
        const int m0 = mt * 16 + laneHi * 8;
        #pragma unroll
        for (int r = 0; r < 8; ++r) {
            const int m = m0 + r;              // agent-row = 3*sample + agent
            const int s3 = m / 3, n3 = m - 3 * s3;
            float y = sAttn[m] * acc[r];
            sVall[(size_t)s3 * 64 + n3 * 16 + ln] = (_Float16)y;
        }
    }

    // ---- P5: GEMM1  v_all[128x64] x Wc1^T[64x256] -> x1 (lrelu, f16) ----
    // two N-tiles in flight -> two independent WMMA chains interleave
    for (int mtl = 0; mtl < 2; ++mtl) {
        const int mt = wave * 2 + mtl;
        const _Float16* arow = sVall + (size_t)(mt * 16 + ln) * 64;
        v16h A0 = lds_frag(arow + laneHi * 16,      arow + laneHi * 16 + 8);
        v16h A1 = lds_frag(arow + 32 + laneHi * 16, arow + 32 + laneHi * 16 + 8);
        for (int j = 0; j < 16; j += 2) {
            v8f acc0, acc1;
            {
                float b0 = bc1[j * 16 + ln], b1 = bc1[(j + 1) * 16 + ln];
                #pragma unroll
                for (int r = 0; r < 8; ++r) { acc0[r] = b0; acc1[r] = b1; }
            }
            const _Float16* br0 = sW1 + (size_t)(j * 16 + ln) * 64;
            const _Float16* br1 = sW1 + (size_t)((j + 1) * 16 + ln) * 64;
            v16h B00 = lds_frag(br0 + laneHi * 16,      br0 + laneHi * 16 + 8);
            v16h B01 = lds_frag(br0 + 32 + laneHi * 16, br0 + 32 + laneHi * 16 + 8);
            v16h B10 = lds_frag(br1 + laneHi * 16,      br1 + laneHi * 16 + 8);
            v16h B11 = lds_frag(br1 + 32 + laneHi * 16, br1 + 32 + laneHi * 16 + 8);
            acc0 = WMMA_F16(A0, B00, acc0);
            acc1 = WMMA_F16(A0, B10, acc1);
            acc0 = WMMA_F16(A1, B01, acc0);
            acc1 = WMMA_F16(A1, B11, acc1);
            const int m0 = mt * 16 + laneHi * 8;
            #pragma unroll
            for (int r = 0; r < 8; ++r) {
                sX1[(size_t)(m0 + r) * 256 + j * 16 + ln]        = (_Float16)lrelu(acc0[r]);
                sX1[(size_t)(m0 + r) * 256 + (j + 1) * 16 + ln]  = (_Float16)lrelu(acc1[r]);
            }
        }
    }

    // ---- P6: GEMM2  x1[128x256] x Wc2^T[256x256], fused lrelu + Wc dot ----
    // two N-tiles in flight: independent accumulator chains, shared A frags
    const float bcv = bc[0];
    for (int mtl = 0; mtl < 2; ++mtl) {
        const int mt = wave * 2 + mtl;
        const _Float16* arow = sX1 + (size_t)(mt * 16 + ln) * 256;
        v16h Af[8];
        #pragma unroll
        for (int k = 0; k < 8; ++k) {
            const int off = 32 * k + laneHi * 16;
            Af[k] = lds_frag(arow + off, arow + off + 8);
        }

        float part[8];
        #pragma unroll
        for (int r = 0; r < 8; ++r) part[r] = 0.f;

        for (int j = 0; j < 16; j += 2) {
            v8f acc0, acc1;
            {
                float b0 = bc2[j * 16 + ln], b1 = bc2[(j + 1) * 16 + ln];
                #pragma unroll
                for (int r = 0; r < 8; ++r) { acc0[r] = b0; acc1[r] = b1; }
            }
            const _Float16* br0 = sW2 + (size_t)(j * 16 + ln) * 256;
            const _Float16* br1 = sW2 + (size_t)((j + 1) * 16 + ln) * 256;
            #pragma unroll
            for (int k = 0; k < 8; ++k) {
                const int off = 32 * k + laneHi * 16;
                v16h B0 = lds_frag(br0 + off, br0 + off + 8);
                v16h B1 = lds_frag(br1 + off, br1 + off + 8);
                acc0 = WMMA_F16(Af[k], B0, acc0);
                acc1 = WMMA_F16(Af[k], B1, acc1);
            }
            const float wc0 = sWc[j * 16 + ln];
            const float wc1 = sWc[(j + 1) * 16 + ln];
            #pragma unroll
            for (int r = 0; r < 8; ++r) {
                part[r] = fmaf(lrelu(acc0[r]), wc0, part[r]);
                part[r] = fmaf(lrelu(acc1[r]), wc1, part[r]);
            }
        }

        #pragma unroll
        for (int mask = 1; mask <= 8; mask <<= 1)
            #pragma unroll
            for (int r = 0; r < 8; ++r)
                part[r] += __shfl_xor(part[r], mask, 32);

        if (ln == 0) {
            const size_t base = (size_t)blockIdx.x * TB + mt * 16 + laneHi * 8;
            #pragma unroll
            for (int r = 0; r < 8; ++r) out[base + r] = part[r] + bcv;
        }
    }
}

// ---------------------------------------------------------------------------
extern "C" void kernel_launch(void* const* d_in, const int* in_sizes, int n_in,
                              void* d_out, int out_size, void* d_ws, size_t ws_size,
                              hipStream_t stream) {
    const float* obs   = (const float*)d_in[0];
    const float* act   = (const float*)d_in[1];
    const float* W_mlp = (const float*)d_in[2];
    const float* b_mlp = (const float*)d_in[3];
    const float* Wq    = (const float*)d_in[4];
    const float* bq    = (const float*)d_in[5];
    const float* Wk    = (const float*)d_in[6];
    const float* bk    = (const float*)d_in[7];
    const float* Wv    = (const float*)d_in[8];
    const float* bv    = (const float*)d_in[9];
    const float* Wa    = (const float*)d_in[10];
    const float* ba    = (const float*)d_in[11];
    const float* Wc1   = (const float*)d_in[12];
    const float* bc1   = (const float*)d_in[13];
    const float* Wc2   = (const float*)d_in[14];
    const float* bc2   = (const float*)d_in[15];
    const float* Wc    = (const float*)d_in[16];
    const float* bc    = (const float*)d_in[17];

    const int Bn = in_sizes[0] / 48;

    // d_ws carve (halves, then f32 region)
    _Float16* w2h   = (_Float16*)d_ws;          // 65536
    _Float16* w1h   = w2h + 65536;              // 16384
    _Float16* wmlph = w1h + 16384;              // 1024
    _Float16* wvh   = wmlph + 1024;             // 512
    float*    wka   = (float*)(wvh + 512);      // 32
    float*    wqa   = wka + 32;                 // 32
    float*    consta = wqa + 32;                // 1

    critic_prep<<<(256 * 256 + 255) / 256, 256, 0, stream>>>(
        Wc2, Wc1, W_mlp, Wv, Wk, Wq, Wa, bk, bq, ba,
        w2h, w1h, wmlph, wvh, wka, wqa, consta);

    const size_t lds_bytes =
        (size_t)(256 * 256 + 256 * 64 + 32 * 32 + 16 * 32 +
                 384 * 32 + 384 * 32 + TB * 64 + TB * 256) * sizeof(_Float16) +
        (384 + 256) * sizeof(float);            // 300544 B < 320 KB

    critic_attention_fused<<<Bn / TB, TB, lds_bytes, stream>>>(
        obs, act, b_mlp, bv, bc1, bc2, Wc, bc, wka, wqa, consta,
        w2h, w1h, wmlph, wvh, (float*)d_out);
}
